// GNNStream_81200651698306
// MI455X (gfx1250) — compile-verified, matrix operand order checked by reading, not run on hardware
//
#include <hip/hip_runtime.h>
#include <hip/hip_bf16.h>

// ---------------------------------------------------------------------------
// GNN-stream transformer for MI455X (gfx1250).
// All GEMMs: v_wmma_f32_16x16x32_bf16, double-buffered LDS tiles staged with
// gfx1250 async global->LDS DMA (ASYNCcnt) when the toolchain exposes it.
// ---------------------------------------------------------------------------

typedef __attribute__((ext_vector_type(16))) __bf16 v16bf;
typedef __attribute__((ext_vector_type(8)))  float  v8f;
typedef int async_i4 __attribute__((vector_size(16)));   // builtin's pointee type

#define GBM 128
#define GBN 128
#define GBK 64

#if __has_builtin(__builtin_amdgcn_global_load_async_to_lds_b128)
#define HAVE_ASYNC_LDS 1
#endif

__device__ __forceinline__ unsigned short f32_to_bf16(float f) {
    unsigned int u = __float_as_uint(f);
    u += 0x7FFFu + ((u >> 16) & 1u);   // round-to-nearest-even
    return (unsigned short)(u >> 16);
}

// 16-byte global -> LDS stage (async DMA on gfx1250 toolchains that have it)
__device__ __forceinline__ void stage_b128(unsigned short* ldst,
                                           const unsigned short* gsrc) {
#ifdef HAVE_ASYNC_LDS
    __builtin_amdgcn_global_load_async_to_lds_b128(
        (__attribute__((address_space(1))) async_i4*)gsrc,
        (__attribute__((address_space(3))) async_i4*)ldst, 0, 0);
#else
    *(uint4*)ldst = *(const uint4*)gsrc;
#endif
}

__device__ __forceinline__ void wait_stage() {
#ifdef HAVE_ASYNC_LDS
#if __has_builtin(__builtin_amdgcn_s_wait_asynccnt)
    __builtin_amdgcn_s_wait_asynccnt(0);
#else
    asm volatile("s_wait_asynccnt 0x0" ::: "memory");
#endif
#endif
}

// ---------------------------------------------------------------------------
// fp32 -> bf16 conversion (grid-stride)
// ---------------------------------------------------------------------------
__global__ void f2bf_kernel(const float* __restrict__ src,
                            unsigned short* __restrict__ dst, long long n) {
    long long i = (long long)blockIdx.x * blockDim.x + threadIdx.x;
    long long stride = (long long)gridDim.x * blockDim.x;
    for (; i < n; i += stride) dst[i] = f32_to_bf16(src[i]);
}

// ---------------------------------------------------------------------------
// fp32 (K x N) -> bf16 transposed (N x K), 32x32 LDS tiles, 256 threads
// ---------------------------------------------------------------------------
__global__ void __launch_bounds__(256)
f2bf_t_kernel(const float* __restrict__ src, unsigned short* __restrict__ dst,
              int K, int N) {
    __shared__ float tile[32][33];
    const int kt = blockIdx.y * 32, nt = blockIdx.x * 32;
    const int tx = threadIdx.x & 31, ty = threadIdx.x >> 5;   // 32 x 8
    #pragma unroll
    for (int i = ty; i < 32; i += 8)
        tile[i][tx] = src[(long long)(kt + i) * N + nt + tx];
    __syncthreads();
    #pragma unroll
    for (int i = ty; i < 32; i += 8)
        dst[(long long)(nt + i) * K + kt + tx] = f32_to_bf16(tile[tx][i]);
}

// ---------------------------------------------------------------------------
// Tiled WMMA GEMM (NT form):  C[b] = alpha * (A[b] @ B[b]^T + bias)
//   A: M x K, row stride lda (bf16/u16).  B: N x K, row stride ldb.
//   Cf (f32, row-major N) and/or Cb (bf16): cbT=0 -> row-major (ldcb),
//   cbT=1 -> transposed store Cb[col*ldcb + row].
//   Tile 128x128x64, double-buffered LDS, async staged; 8 waves;
//   wave grid 4(m) x 2(n); 16 WMMAs per k-tile per wave.
// ---------------------------------------------------------------------------
__global__ void __launch_bounds__(256)
wmma_gemm_nt_kernel(const unsigned short* __restrict__ A,
                    const unsigned short* __restrict__ B,
                    const float* __restrict__ bias,
                    float* __restrict__ Cf,
                    unsigned short* __restrict__ Cb,
                    int M, int N, int K,
                    long long lda, long long ldb, long long ldcb,
                    long long sA, long long sB, long long sC,
                    float alpha, int cbT)
{
    __shared__ __align__(128) unsigned short As[2][GBM * GBK];   // 16 KB x2
    __shared__ __align__(128) unsigned short Bs[2][GBN * GBK];   // 16 KB x2

    const int tid    = threadIdx.x;
    const int lane   = tid & 31;
    const int wid    = tid >> 5;
    const int wave_m = wid >> 1;   // 0..3 -> 32-row strip
    const int wave_n = wid & 1;    // 0..1 -> 64-col strip
    const int lrow   = lane & 15;
    const int lhalf  = lane >> 4;

    const int m0 = blockIdx.y * GBM;
    const int n0 = blockIdx.x * GBN;
    const int b  = blockIdx.z;

    const unsigned short* Ab = A + (long long)b * sA + (long long)m0 * lda;
    const unsigned short* Bb = B + (long long)b * sB + (long long)n0 * ldb;

    v8f acc[2][4] = {};

    auto stage = [&](int kt, int buf) {
        const unsigned short* sa = Ab + (long long)kt * GBK;
        const unsigned short* sb = Bb + (long long)kt * GBK;
        #pragma unroll
        for (int i = 0; i < 4; ++i) {
            int u   = tid * 4 + i;        // 0..1023 (1024 x b128 = 16 KB)
            int row = u >> 3;             // 0..127
            int seg = (u & 7) * 8;        // 0,8,...,56
            stage_b128(&As[buf][row * GBK + seg], sa + (long long)row * lda + seg);
            stage_b128(&Bs[buf][row * GBK + seg], sb + (long long)row * ldb + seg);
        }
    };

    stage(0, 0);
    const int KT = K / GBK;
    for (int kt = 0; kt < KT; ++kt) {
        const int cur = kt & 1;
        wait_stage();          // my async DMAs for buffer `cur` have landed
        __syncthreads();       // everyone's DMAs landed; prior reads retired
        if (kt + 1 < KT) stage(kt + 1, cur ^ 1);

        #pragma unroll
        for (int ks = 0; ks < 2; ++ks) {
            #pragma unroll
            for (int mi = 0; mi < 2; ++mi) {
                v16bf af = *(const v16bf*)&As[cur][(wave_m * 32 + mi * 16 + lrow) * GBK + ks * 32 + lhalf * 16];
                #pragma unroll
                for (int ni = 0; ni < 4; ++ni) {
                    v16bf bf = *(const v16bf*)&Bs[cur][(wave_n * 64 + ni * 16 + lrow) * GBK + ks * 32 + lhalf * 16];
                    acc[mi][ni] = __builtin_amdgcn_wmma_f32_16x16x32_bf16(
                        false, af, false, bf, (short)0, acc[mi][ni], false, false);
                }
            }
        }
    }

    // ---- epilogue: v = alpha * (acc + bias) ----
    float*          Cfb = Cf ? (Cf + (long long)b * sC) : nullptr;
    unsigned short* Cbb = Cb ? (Cb + (long long)b * sC) : nullptr;
    #pragma unroll
    for (int mi = 0; mi < 2; ++mi) {
        #pragma unroll
        for (int ni = 0; ni < 4; ++ni) {
            int col  = n0 + wave_n * 64 + ni * 16 + lrow;
            float bv = bias ? bias[col] : 0.0f;
            #pragma unroll
            for (int r = 0; r < 8; ++r) {
                int row = m0 + wave_m * 32 + mi * 16 + lhalf * 8 + r;
                float v = (acc[mi][ni][r] + bv) * alpha;
                if (Cfb) Cfb[(long long)row * N + col] = v;
                if (Cbb) {
                    long long idx = cbT ? ((long long)col * ldcb + row)
                                        : ((long long)row * ldcb + col);
                    Cbb[idx] = f32_to_bf16(v);
                }
            }
        }
    }
}

// ---------------------------------------------------------------------------
// Row LayerNorm + ReLU.  One block per row (N = 512 or 256).
// ---------------------------------------------------------------------------
__global__ void __launch_bounds__(256)
ln_relu_kernel(const float* __restrict__ X, const float* __restrict__ g,
               const float* __restrict__ bb, float* __restrict__ outF,
               unsigned short* __restrict__ outB, int N)
{
    __shared__ float red[256];
    const long long row = blockIdx.x;
    const float* x = X + row * N;
    const int tid = threadIdx.x;

    float s = 0.f, ss = 0.f;
    for (int c = tid; c < N; c += 256) { float v = x[c]; s += v; ss += v * v; }
    red[tid] = s; __syncthreads();
    for (int k = 128; k > 0; k >>= 1) { if (tid < k) red[tid] += red[tid + k]; __syncthreads(); }
    float mu = red[0] / (float)N; __syncthreads();
    red[tid] = ss; __syncthreads();
    for (int k = 128; k > 0; k >>= 1) { if (tid < k) red[tid] += red[tid + k]; __syncthreads(); }
    float var = red[0] / (float)N - mu * mu;
    float inv = rsqrtf(var + 1e-5f);

    for (int c = tid; c < N; c += 256) {
        float v = (x[c] - mu) * inv * g[c] + bb[c];
        v = fmaxf(v, 0.f);
        if (outF) outF[row * N + c] = v;
        if (outB) outB[row * N + c] = f32_to_bf16(v);
    }
}

// ---------------------------------------------------------------------------
// Fused: mask -> softmax over 1024 keys -> * exp(-dist*|gamma|) -> bf16 attn.
// One block per (query s, batch b); 256 threads x 4 keys each.
// ---------------------------------------------------------------------------
__global__ void __launch_bounds__(256)
attn_softmax_kernel(const float* __restrict__ scores, const float* __restrict__ coords,
                    const unsigned char* __restrict__ mask, const float* __restrict__ gamma,
                    unsigned short* __restrict__ attn)
{
    const int S = 1024;
    const int s = blockIdx.x;
    const int b = blockIdx.y;
    const int tid = threadIdx.x;
    __shared__ float red[256];

    const float ga  = fabsf(gamma[0]);
    const float cxs = coords[((long long)b * S + s) * 3 + 0];
    const float cys = coords[((long long)b * S + s) * 3 + 1];
    const float n2s = cxs * cxs + cys * cys;

    const float*         srow = scores + ((long long)b * S + s) * S;
    const unsigned char* mrow = mask + (long long)b * S;

    float v[4], adj[4];
    float lmax = -3.0e38f;
    #pragma unroll
    for (int i = 0; i < 4; ++i) {
        int t = tid + i * 256;
        float cxt = coords[((long long)b * S + t) * 3 + 0];
        float cyt = coords[((long long)b * S + t) * 3 + 1];
        float d2  = n2s + cxt * cxt + cyt * cyt - 2.f * (cxs * cxt + cys * cyt);
        float dist = sqrtf(fmaxf(d2, 0.f));
        adj[i] = __expf(-dist * ga);
        float sc = mrow[t] ? -1.0e9f : srow[t];
        v[i] = sc;
        lmax = fmaxf(lmax, sc);
    }
    red[tid] = lmax; __syncthreads();
    for (int k = 128; k > 0; k >>= 1) { if (tid < k) red[tid] = fmaxf(red[tid], red[tid + k]); __syncthreads(); }
    float m = red[0]; __syncthreads();

    float lsum = 0.f;
    #pragma unroll
    for (int i = 0; i < 4; ++i) { v[i] = __expf(v[i] - m); lsum += v[i]; }
    red[tid] = lsum; __syncthreads();
    for (int k = 128; k > 0; k >>= 1) { if (tid < k) red[tid] += red[tid + k]; __syncthreads(); }
    float invsum = 1.f / red[0];

    unsigned short* arow = attn + ((long long)b * S + s) * S;
    #pragma unroll
    for (int i = 0; i < 4; ++i) {
        int t = tid + i * 256;
        arow[t] = f32_to_bf16(v[i] * invsum * adj[i]);
    }
}

// ---------------------------------------------------------------------------
// Masked mean pool over S:  pooled[b,d] = sum_s out*~mask / max(cnt,1e-9)
// ---------------------------------------------------------------------------
__global__ void __launch_bounds__(256)
pool_kernel(const float* __restrict__ out, const unsigned char* __restrict__ mask,
            float* __restrict__ pooled, float* __restrict__ pooled_out)
{
    const int S = 1024, D = 256;
    const int b = blockIdx.x;
    const int d = threadIdx.x;
    const unsigned char* mrow = mask + (long long)b * S;
    const float*         orow = out + (long long)b * S * D;
    float sum = 0.f, cnt = 0.f;
    for (int s = 0; s < S; ++s) {
        float w = mrow[s] ? 0.f : 1.f;
        cnt += w;
        sum += orow[(long long)s * D + d] * w;
    }
    float p = sum / fmaxf(cnt, 1e-9f);
    pooled[b * D + d]     = p;
    pooled_out[b * D + d] = p;
}

// ---------------------------------------------------------------------------
// Tiny classifier: relu(pooled @ W1 + b1) @ W2 + b2  (256->128->400)
// ---------------------------------------------------------------------------
__global__ void __launch_bounds__(256)
classify_kernel(const float* __restrict__ pooled,
                const float* __restrict__ w1, const float* __restrict__ b1,
                const float* __restrict__ w2, const float* __restrict__ b2,
                float* __restrict__ logits)
{
    const int D = 256, H = 128, C = 400;
    __shared__ float pin[256];
    __shared__ float hid[128];
    const int b = blockIdx.x;
    const int tid = threadIdx.x;
    pin[tid] = pooled[b * D + tid];
    __syncthreads();
    if (tid < H) {
        float acc = b1[tid];
        for (int i = 0; i < D; ++i) acc += pin[i] * w1[i * H + tid];
        hid[tid] = fmaxf(acc, 0.f);
    }
    __syncthreads();
    for (int j = tid; j < C; j += 256) {
        float acc = b2[j];
        for (int i = 0; i < H; ++i) acc += hid[i] * w2[i * C + j];
        logits[b * C + j] = acc;
    }
}

// ---------------------------------------------------------------------------
extern "C" void kernel_launch(void* const* d_in, const int* in_sizes, int n_in,
                              void* d_out, int out_size, void* d_ws, size_t ws_size,
                              hipStream_t stream)
{
    const int B = 32, S = 1024, CLIP = 512, ND = 512, BD = 256, NC = 400;
    const long long BS = (long long)B * S;   // 32768 rows

    const float*         local_feat = (const float*)d_in[0];
    const float*         coords     = (const float*)d_in[1];
    const unsigned char* mask       = (const unsigned char*)d_in[2];  // jnp bool = 1 byte
    const float* enc_w  = (const float*)d_in[3];
    const float* enc_b  = (const float*)d_in[4];
    const float* enc_g  = (const float*)d_in[5];
    const float* enc_bb = (const float*)d_in[6];
    const float* gamma  = (const float*)d_in[7];
    const float* wq_w   = (const float*)d_in[8];
    const float* wq_b   = (const float*)d_in[9];
    const float* wk_w   = (const float*)d_in[10];
    const float* wk_b   = (const float*)d_in[11];
    const float* wv_w   = (const float*)d_in[12];
    const float* wv_b   = (const float*)d_in[13];
    const float* bot_w  = (const float*)d_in[14];
    const float* bot_b  = (const float*)d_in[15];
    const float* bot_g  = (const float*)d_in[16];
    const float* bot_bb = (const float*)d_in[17];
    const float* cls1_w = (const float*)d_in[18];
    const float* cls1_b = (const float*)d_in[19];
    const float* cls2_w = (const float*)d_in[20];
    const float* cls2_b = (const float*)d_in[21];
    (void)in_sizes; (void)n_in; (void)out_size; (void)ws_size;

    char* ws = (char*)d_ws;
    size_t off = 0;
    auto alloc = [&](size_t bytes) { size_t o = off; off += (bytes + 255) & ~(size_t)255; return o; };

    unsigned short* Xb      = (unsigned short*)(ws + alloc((size_t)BS * CLIP * 2));
    unsigned short* WencT   = (unsigned short*)(ws + alloc((size_t)CLIP * ND * 2));  // ND x CLIP
    unsigned short* WqT     = (unsigned short*)(ws + alloc((size_t)ND * ND * 2));    // ND x ND
    unsigned short* WkT     = (unsigned short*)(ws + alloc((size_t)ND * ND * 2));
    unsigned short* WvT     = (unsigned short*)(ws + alloc((size_t)ND * ND * 2));
    unsigned short* WbotT   = (unsigned short*)(ws + alloc((size_t)ND * BD * 2));    // BD x ND
    float*          preF    = (float*)(ws + alloc((size_t)BS * ND * 4));             // reused enc & bot
    unsigned short* Hb      = (unsigned short*)(ws + alloc((size_t)BS * ND * 2));
    unsigned short* Qb      = (unsigned short*)(ws + alloc((size_t)BS * ND * 2));
    unsigned short* Kb      = (unsigned short*)(ws + alloc((size_t)BS * ND * 2));
    unsigned short* VbT     = (unsigned short*)(ws + alloc((size_t)BS * ND * 2));    // ND x BS
    float*          scoresF = (float*)(ws + alloc((size_t)B * S * S * 4));
    unsigned short* attnB   = (unsigned short*)(ws + alloc((size_t)B * S * S * 2));
    unsigned short* ctxB    = (unsigned short*)(ws + alloc((size_t)BS * ND * 2));
    float*          outF    = (float*)(ws + alloc((size_t)BS * BD * 4));
    float*          pooledF = (float*)(ws + alloc((size_t)B * BD * 4));

    float* logits     = (float*)d_out;            // (32, 400)
    float* pooled_out = (float*)d_out + B * NC;   // (32, 256)

    // fp32 -> bf16 packing; weights packed transposed (N x K) for NT GEMMs
    f2bf_kernel<<<1024, 256, 0, stream>>>(local_feat, Xb, BS * CLIP);
    f2bf_t_kernel<<<dim3(ND / 32, CLIP / 32), 256, 0, stream>>>(enc_w, WencT, CLIP, ND);
    f2bf_t_kernel<<<dim3(ND / 32, ND / 32), 256, 0, stream>>>(wq_w, WqT, ND, ND);
    f2bf_t_kernel<<<dim3(ND / 32, ND / 32), 256, 0, stream>>>(wk_w, WkT, ND, ND);
    f2bf_t_kernel<<<dim3(ND / 32, ND / 32), 256, 0, stream>>>(wv_w, WvT, ND, ND);
    f2bf_t_kernel<<<dim3(BD / 32, ND / 32), 256, 0, stream>>>(bot_w, WbotT, ND, BD);

    const float qscale = 0.044194173824159216f;   // 1/sqrt(512)

    // encoder: preF = X @ enc_w + enc_b ; h = bf16(relu(LN(preF)))
    wmma_gemm_nt_kernel<<<dim3(ND / GBN, BS / GBM, 1), 256, 0, stream>>>(
        Xb, WencT, enc_b, preF, nullptr, (int)BS, ND, CLIP,
        CLIP, CLIP, 0, 0, 0, 0, 1.0f, 0);
    ln_relu_kernel<<<(int)BS, 256, 0, stream>>>(preF, enc_g, enc_bb, nullptr, Hb, ND);

    // q (pre-scaled by 1/sqrt(ND)), k, v (v stored transposed ND x BS)
    wmma_gemm_nt_kernel<<<dim3(ND / GBN, BS / GBM, 1), 256, 0, stream>>>(
        Hb, WqT, wq_b, nullptr, Qb, (int)BS, ND, ND, ND, ND, ND, 0, 0, 0, qscale, 0);
    wmma_gemm_nt_kernel<<<dim3(ND / GBN, BS / GBM, 1), 256, 0, stream>>>(
        Hb, WkT, wk_b, nullptr, Kb, (int)BS, ND, ND, ND, ND, ND, 0, 0, 0, 1.0f, 0);
    wmma_gemm_nt_kernel<<<dim3(ND / GBN, BS / GBM, 1), 256, 0, stream>>>(
        Hb, WvT, wv_b, nullptr, VbT, (int)BS, ND, ND, ND, ND, BS, 0, 0, 0, 1.0f, 1);

    // scores[b] = q[b] @ k[b]^T   (scale folded into q)
    wmma_gemm_nt_kernel<<<dim3(S / GBN, S / GBM, B), 256, 0, stream>>>(
        Qb, Kb, nullptr, scoresF, nullptr, S, S, ND,
        ND, ND, 0, (long long)S * ND, (long long)S * ND, (long long)S * S, 1.0f, 0);

    // mask + softmax + spatial adjacency -> bf16 attn
    attn_softmax_kernel<<<dim3(S, B), 256, 0, stream>>>(scoresF, coords, mask, gamma, attnB);

    // ctx[b] = attn[b] @ v[b] : B = VbT (ND rows x BS cols), batch col offset
    wmma_gemm_nt_kernel<<<dim3(ND / GBN, S / GBM, B), 256, 0, stream>>>(
        attnB, VbT, nullptr, nullptr, ctxB, S, ND, S,
        S, BS, ND, (long long)S * S, (long long)S, (long long)S * ND, 1.0f, 0);

    // bottleneck: preF = ctx @ bot_w + bot_b ; out = relu(LN(preF))
    wmma_gemm_nt_kernel<<<dim3(BD / GBN, BS / GBM, 1), 256, 0, stream>>>(
        ctxB, WbotT, bot_b, preF, nullptr, (int)BS, BD, ND,
        ND, ND, 0, 0, 0, 0, 1.0f, 0);
    ln_relu_kernel<<<(int)BS, 256, 0, stream>>>(preF, bot_g, bot_bb, outF, nullptr, BD);

    // masked mean pool + classifier
    pool_kernel<<<B, BD, 0, stream>>>(outF, mask, pooledF, pooled_out);
    classify_kernel<<<B, 256, 0, stream>>>(pooledF, cls1_w, cls1_b, cls2_w, cls2_b, logits);
}
